// SpatiotemporalAttention_9363028706002
// MI455X (gfx1250) — matrix operand order repaired and checked by reading, not compiled, not run on hardware
//
#include <hip/hip_runtime.h>

typedef __attribute__((ext_vector_type(16))) _Float16 v16h;
typedef __attribute__((ext_vector_type(8)))  _Float16 v8h;
typedef __attribute__((ext_vector_type(8)))  float    v8f;
typedef __attribute__((ext_vector_type(4)))  float    v4f;
typedef __attribute__((ext_vector_type(4)))  int      v4i;

namespace sta {
constexpr int   kB   = 4;
constexpr int   kC   = 256;
constexpr int   kCI  = 128;
constexpr int   kN   = 2304;            // 48*48
constexpr float kEPS = 1e-5f;
constexpr long  kBS  = (long)kB * kCI * kN;   // branch stride in Yh
}
using namespace sta;

// ---------------------------------------------------------------------------
// CDNA5 async global->LDS (ASYNCcnt) with safe fallback.
// Builtin signature (from compiler diagnostic): first param is v4i in the
// global ("__device__") address space; LDS destination second.
// ---------------------------------------------------------------------------
#if defined(__has_builtin)
#if __has_builtin(__builtin_amdgcn_global_load_async_to_lds_b128)
#define STA_ASYNC_LDS 1
#endif
#endif

__device__ __forceinline__ void sta_copy16_g2l(const _Float16* g, _Float16* l) {
#ifdef STA_ASYNC_LDS
  __builtin_amdgcn_global_load_async_to_lds_b128(
      (__attribute__((address_space(1))) v4i*)g,
      (__attribute__((address_space(3))) v4i*)l, 0, 0);
#else
  *(v8h*)l = *(const v8h*)g;
#endif
}

__device__ __forceinline__ void sta_wait_async() {
#ifdef STA_ASYNC_LDS
#if __has_builtin(__builtin_amdgcn_s_wait_asynccnt)
  __builtin_amdgcn_s_wait_asynccnt(0);
#else
  asm volatile("s_wait_asynccnt 0x0" ::: "memory");
#endif
#endif
}

// ---------------------------------------------------------------------------
// Fold eval-mode BN into 1x1 conv weights.
// pre:  y = W @ BN(x) + b  ->  y = (W*s) @ x + (b + W@(beta - mean*s))
// ---------------------------------------------------------------------------
__global__ void fold_pre_kernel(const float* __restrict__ bn_pre,
                                const float* __restrict__ w_pre,
                                const float* __restrict__ b_pre,
                                _Float16* __restrict__ Wh,
                                float* __restrict__ beff) {
  int t = blockIdx.x * blockDim.x + threadIdx.x;
  if (t >= 4 * kCI) return;
  int i = t / kCI, o = t % kCI;
  const float* g  = bn_pre + (i * 4 + 0) * kC;
  const float* be = bn_pre + (i * 4 + 1) * kC;
  const float* mu = bn_pre + (i * 4 + 2) * kC;
  const float* va = bn_pre + (i * 4 + 3) * kC;
  const float* w  = w_pre + (long)(i * kCI + o) * kC;
  _Float16* wh    = Wh + (long)(i * kCI + o) * kC;
  float acc = b_pre[i * kCI + o];
  for (int c = 0; c < kC; ++c) {
    float s = g[c] * rsqrtf(va[c] + kEPS);
    wh[c] = (_Float16)(w[c] * s);
    acc += w[c] * (be[c] - mu[c] * s);
  }
  beff[i * kCI + o] = acc;
}

// post: BN(W @ y + b) -> (s*W) @ y + ((b - mean)*s + beta)
__global__ void fold_post_kernel(const float* __restrict__ bn_post,
                                 const float* __restrict__ w_post,
                                 const float* __restrict__ b_post,
                                 _Float16* __restrict__ Wh,
                                 float* __restrict__ beff) {
  int t = blockIdx.x * blockDim.x + threadIdx.x;
  if (t >= 2 * kC) return;
  int j = t / kC, o = t % kC;
  float s    = bn_post[(j * 4 + 0) * kC + o] * rsqrtf(bn_post[(j * 4 + 3) * kC + o] + kEPS);
  float beta = bn_post[(j * 4 + 1) * kC + o];
  float mu   = bn_post[(j * 4 + 2) * kC + o];
  const float* w = w_post + (long)(j * kC + o) * kCI;
  _Float16* wh   = Wh + (long)(j * kC + o) * kCI;
  for (int ci = 0; ci < kCI; ++ci) wh[ci] = (_Float16)(w[ci] * s);
  beff[j * kC + o] = (b_post[j * kC + o] - mu) * s + beta;
}

__global__ void f2h_kernel(const float* __restrict__ src, _Float16* __restrict__ dst, int n) {
  int i = blockIdx.x * 256 + threadIdx.x;
  if (i < n) dst[i] = (_Float16)src[i];
}

// ---------------------------------------------------------------------------
// Tile staging: A -> As[m][k], B -> Bs[n][k] (transposed for clean B frags).
// Contiguous 16B paths (TRANS_A==0 A, BMODE==3 B) use async global->LDS.
// ---------------------------------------------------------------------------
template <int TRANS_A, int BMODE>
__device__ __forceinline__ void stage_tile(
    _Float16 (*__restrict__ As)[40], _Float16 (*__restrict__ Bs)[40],
    const _Float16* __restrict__ Ab, int lda,
    const _Float16* __restrict__ Bh, const float* __restrict__ Bf, int ldb,
    const float* __restrict__ smx, const float* __restrict__ siv,
    int bm, int bn, int k0, int tid) {
  __builtin_prefetch((const void*)(Ab + (long)bm * (TRANS_A ? 1 : lda) + k0 + 32), 0, 1);
#pragma unroll
  for (int i = 0; i < 2; ++i) {
    int cid = tid + i * 256;                    // 0..511 -> 128x32 halves
    if (TRANS_A == 0) {
      int row = cid >> 2;
      int kc  = (cid & 3) << 3;
      sta_copy16_g2l(Ab + (long)(bm + row) * lda + (k0 + kc), &As[row][kc]);
    } else {
      int kk = cid >> 4;
      int mc = (cid & 15) << 3;
      v8h v = *(const v8h*)(Ab + (long)(k0 + kk) * lda + (bm + mc));
#pragma unroll
      for (int j = 0; j < 8; ++j) As[mc + j][kk] = v[j];
    }
  }
#pragma unroll
  for (int i = 0; i < 2; ++i) {
    int cid = tid + i * 256;
    if (BMODE == 0) {
      int kk = cid >> 4;
      int nc = (cid & 15) << 3;
      v8h v = *(const v8h*)(Bh + (long)(k0 + kk) * ldb + (bn + nc));
#pragma unroll
      for (int j = 0; j < 8; ++j) Bs[nc + j][kk] = v[j];
    } else if (BMODE == 1) {
      int kk = cid >> 4;
      int nc = (cid & 15) << 3;
      const float* src = Bf + (long)(k0 + kk) * ldb + (bn + nc);
      v4f v0 = *(const v4f*)src;
      v4f v1 = *(const v4f*)(src + 4);
#pragma unroll
      for (int j = 0; j < 4; ++j) {
        int n0 = bn + nc + j;
        int n1 = n0 + 4;
        Bs[nc + j][kk]     = (_Float16)(__expf(v0[j] - smx[n0]) * siv[n0]);
        Bs[nc + 4 + j][kk] = (_Float16)(__expf(v1[j] - smx[n1]) * siv[n1]);
      }
    } else if (BMODE == 2) {
      int mm = cid >> 2;
      int kc = (cid & 3) << 3;
      const float* src = Bf + (long)(bn + mm) * ldb + (k0 + kc);
      float mx = smx[bn + mm], iv = siv[bn + mm];
      v4f v0 = *(const v4f*)src;
      v4f v1 = *(const v4f*)(src + 4);
      v8h o;
#pragma unroll
      for (int j = 0; j < 4; ++j) {
        o[j]     = (_Float16)(__expf(v0[j] - mx) * iv);
        o[j + 4] = (_Float16)(__expf(v1[j] - mx) * iv);
      }
      *(v8h*)&Bs[mm][kc] = o;
    } else {  // BMODE == 3 (f16 transposed source, contiguous in k)
      int mm = cid >> 2;
      int kc = (cid & 3) << 3;
      sta_copy16_g2l(Bh + (long)(bn + mm) * ldb + (k0 + kc), &Bs[mm][kc]);
    }
  }
}

// ---------------------------------------------------------------------------
// Generic 128x128-tile WMMA GEMM, 256 threads = 8 waves (2x4), wave = 64x32.
// Double-buffered LDS: stage slab k+1 (async where possible) while WMMAs
// consume slab k; single barrier per K-step.
// A is f16 (TRANS_A=0: [M][K] row-major; TRANS_A=1: stored [K][M]).
// BMODE: 0 = f16 [K][N];  1 = f32 Es[K(=n)][N(=m)] with col-softmax stats by m;
//        2 = f32 Es transposed: B[k][m] = exp(Es[m][k]-rmax[m])*rinv[m];
//        3 = f16 transposed:    B[k][n] = G[n][k].
// EPI:   0 = f16 store; 1 = f32 store; 2 = bias + residual + f32 store.
// ---------------------------------------------------------------------------
template <int TRANS_A, int BMODE, int EPI>
__global__ __launch_bounds__(256, 2)
void gemm_wmma_f16(const _Float16* __restrict__ A, long sA, int lda,
                   const void* __restrict__ Bsrc, long sB, int ldb,
                   const float* __restrict__ smax, const float* __restrict__ sinv,
                   const float* __restrict__ bias,
                   const float* __restrict__ resid, long sR,
                   void* __restrict__ Cout, long sC, int ldc,
                   int Nn, int K) {
  __shared__ __align__(16) _Float16 As[2][128][40];
  __shared__ __align__(16) _Float16 Bs[2][128][40];

  const int tid  = threadIdx.x;
  const int lane = tid & 31;
  const int wave = tid >> 5;
  const int wm   = (wave >> 2) * 64;   // wave row offset in block tile
  const int wn   = (wave & 3) * 32;    // wave col offset in block tile
  const int bm   = blockIdx.y * 128;
  const int bn   = blockIdx.x * 128;
  const int bat  = blockIdx.z;

  const _Float16* Ab = A + (long)bat * sA;
  const _Float16* Bh = (const _Float16*)Bsrc + (long)bat * sB;
  const float*    Bf = (const float*)Bsrc + (long)bat * sB;
  const float* smx = (BMODE == 1 || BMODE == 2) ? smax + (long)bat * Nn : nullptr;
  const float* siv = (BMODE == 1 || BMODE == 2) ? sinv + (long)bat * Nn : nullptr;
  const float* Rb  = (EPI == 2) ? resid + (long)bat * sR : nullptr;

  v8f acc[4][2] = {};

  // prologue: stage first K slab into buffer 0
  stage_tile<TRANS_A, BMODE>(As[0], Bs[0], Ab, lda, Bh, Bf, ldb, smx, siv, bm, bn, 0, tid);
  sta_wait_async();
  __syncthreads();

  for (int k0 = 0; k0 < K; k0 += 32) {
    const int cur = (k0 >> 5) & 1;
    if (k0 + 32 < K)
      stage_tile<TRANS_A, BMODE>(As[cur ^ 1], Bs[cur ^ 1], Ab, lda, Bh, Bf, ldb,
                                 smx, siv, bm, bn, k0 + 32, tid);

    // ---- fragments per documented CDNA5 wave32 layouts ----
    v16h af[4], bfr[2];
#pragma unroll
    for (int i = 0; i < 4; ++i) {
      int row = wm + i * 16 + (lane & 15);
      int ko  = (lane >> 4) << 3;  // A: lane<16 -> K 0-7 & 16-23; lane>=16 -> K 8-15 & 24-31
      v8h lo = *(const v8h*)&As[cur][row][ko];
      v8h hi = *(const v8h*)&As[cur][row][ko + 16];
#pragma unroll
      for (int j = 0; j < 8; ++j) { af[i][j] = lo[j]; af[i][j + 8] = hi[j]; }
    }
#pragma unroll
    for (int jn = 0; jn < 2; ++jn) {
      int row = wn + jn * 16 + (lane & 15);
      int ko  = (lane >> 4) << 3;  // B: lane<16 -> K 0-7 & 16-23; lane>=16 -> K 8-15 & 24-31
      v8h lo = *(const v8h*)&Bs[cur][row][ko];
      v8h hi = *(const v8h*)&Bs[cur][row][ko + 16];
#pragma unroll
      for (int j = 0; j < 8; ++j) { bfr[jn][j] = lo[j]; bfr[jn][j + 8] = hi[j]; }
    }
#pragma unroll
    for (int i = 0; i < 4; ++i)
#pragma unroll
      for (int jn = 0; jn < 2; ++jn)
        acc[i][jn] = __builtin_amdgcn_wmma_f32_16x16x32_f16(
            false, af[i], false, bfr[jn], (short)0, acc[i][jn], false, false);

    sta_wait_async();
    __syncthreads();
  }

  // ---- epilogue (C/D layout: n = lane%16, m = vgpr + 8*(lane>=16)) ----
#pragma unroll
  for (int i = 0; i < 4; ++i) {
#pragma unroll
    for (int jn = 0; jn < 2; ++jn) {
      int n  = bn + wn + jn * 16 + (lane & 15);
      int mb = bm + wm + i * 16 + ((lane >> 4) << 3);
#pragma unroll
      for (int r = 0; r < 8; ++r) {
        int m = mb + r;
        float v = acc[i][jn][r];
        if (bias) v += bias[m];
        long off = (long)m * ldc + n;
        if (EPI == 0) {
          ((_Float16*)Cout)[(long)bat * sC + off] = (_Float16)v;
        } else if (EPI == 1) {
          ((float*)Cout)[(long)bat * sC + off] = v;
        } else {
          ((float*)Cout)[(long)bat * sC + off] = v + Rb[off];
        }
      }
    }
  }
}

// ---------------------------------------------------------------------------
// e_time softmaxes: EG2[c][d] = rowsoftmax(Et)[c][d] (= et1s)
//                   EG1[c][d] = exp(Et[d][c]-colmax_c)/colsum_c (= et2s)
// ---------------------------------------------------------------------------
__global__ void et_softmax_kernel(const float* __restrict__ Et,
                                  _Float16* __restrict__ EG1,
                                  _Float16* __restrict__ EG2) {
  int c = blockIdx.x, b = blockIdx.y, d = threadIdx.x;
  const float* E = Et + (long)b * kCI * kCI;
  float vr = E[c * kCI + d];
  float vc = E[d * kCI + c];
  __shared__ float sh[128];

  sh[d] = vr; __syncthreads();
  for (int o = 64; o; o >>= 1) { if (d < o) sh[d] = fmaxf(sh[d], sh[d + o]); __syncthreads(); }
  float rm = sh[0]; __syncthreads();
  float er = __expf(vr - rm);
  sh[d] = er; __syncthreads();
  for (int o = 64; o; o >>= 1) { if (d < o) sh[d] += sh[d + o]; __syncthreads(); }
  float rs = sh[0]; __syncthreads();

  sh[d] = vc; __syncthreads();
  for (int o = 64; o; o >>= 1) { if (d < o) sh[d] = fmaxf(sh[d], sh[d + o]); __syncthreads(); }
  float cm = sh[0]; __syncthreads();
  float ec = __expf(vc - cm);
  sh[d] = ec; __syncthreads();
  for (int o = 64; o; o >>= 1) { if (d < o) sh[d] += sh[d + o]; __syncthreads(); }
  float cs = sh[0];

  long base = ((long)b * kCI + c) * kCI + d;
  EG2[base] = (_Float16)(er / rs);
  EG1[base] = (_Float16)(ec / cs);
}

// e_space per-column online softmax stats (coalesced across threads)
__global__ void es_colstats_kernel(const float* __restrict__ Es,
                                   float* __restrict__ cmax, float* __restrict__ cinv) {
  int m = blockIdx.x * 256 + threadIdx.x;
  int b = blockIdx.z;
  const float* E = Es + (long)b * kN * kN;
  float mx = -1e30f, s = 0.f;
  for (int n = 0; n < kN; ++n) {
    float v = E[(long)n * kN + m];
    if (v > mx) { s = s * __expf(mx - v) + 1.f; mx = v; }
    else        { s += __expf(v - mx); }
  }
  cmax[(long)b * kN + m] = mx;
  cinv[(long)b * kN + m] = 1.f / s;
}

// e_space per-row online softmax stats (block reduction per row)
__global__ void es_rowstats_kernel(const float* __restrict__ Es,
                                   float* __restrict__ rmax, float* __restrict__ rinv) {
  int n = blockIdx.x, b = blockIdx.z, tid = threadIdx.x;
  const float* row = Es + (long)b * kN * kN + (long)n * kN;
  float mx = -1e30f, s = 0.f;
  for (int m = tid; m < kN; m += 256) {
    float v = row[m];
    if (v > mx) { s = s * __expf(mx - v) + 1.f; mx = v; }
    else        { s += __expf(v - mx); }
  }
  __shared__ float smx[256], ssm[256];
  smx[tid] = mx; ssm[tid] = s; __syncthreads();
  for (int o = 128; o > 0; o >>= 1) {
    if (tid < o) {
      float m2 = smx[tid + o], s2 = ssm[tid + o];
      float nm = fmaxf(smx[tid], m2);
      ssm[tid] = ssm[tid] * __expf(smx[tid] - nm) + s2 * __expf(m2 - nm);
      smx[tid] = nm;
    }
    __syncthreads();
  }
  if (tid == 0) { rmax[(long)b * kN + n] = smx[0]; rinv[(long)b * kN + n] = 1.f / ssm[0]; }
}

// ---------------------------------------------------------------------------
extern "C" void kernel_launch(void* const* d_in, const int* in_sizes, int n_in,
                              void* d_out, int out_size, void* d_ws, size_t ws_size,
                              hipStream_t stream) {
  (void)in_sizes; (void)n_in; (void)out_size; (void)ws_size;
  const float* x1      = (const float*)d_in[0];
  const float* x2      = (const float*)d_in[1];
  const float* bn_pre  = (const float*)d_in[2];
  const float* w_pre   = (const float*)d_in[3];
  const float* b_pre   = (const float*)d_in[4];
  const float* w_post  = (const float*)d_in[5];
  const float* b_post  = (const float*)d_in[6];
  const float* bn_post = (const float*)d_in[7];
  float* out = (float*)d_out;

  char* p = (char*)d_ws;
  auto carve = [&](size_t bytes) -> char* {
    char* r = p; p += (bytes + 255) & ~(size_t)255; return r;
  };
  _Float16* WpreH  = (_Float16*)carve((size_t)4 * kCI * kC * 2);
  float*    bpreF  = (float*)carve((size_t)4 * kCI * 4);
  _Float16* WpostH = (_Float16*)carve((size_t)2 * kC * kCI * 2);
  float*    bpostF = (float*)carve((size_t)2 * kC * 4);
  _Float16* xh     = (_Float16*)carve((size_t)2 * kB * kC * kN * 2);
  _Float16* Yh     = (_Float16*)carve((size_t)4 * kB * kCI * kN * 2);
  float*    Et     = (float*)carve((size_t)kB * kCI * kCI * 4);
  _Float16* EG1h   = (_Float16*)carve((size_t)kB * kCI * kCI * 2);
  _Float16* EG2h   = (_Float16*)carve((size_t)kB * kCI * kCI * 2);
  float*    Es     = (float*)carve((size_t)kB * kN * kN * 4);
  float*    cmaxv  = (float*)carve((size_t)kB * kN * 4);
  float*    cinvv  = (float*)carve((size_t)kB * kN * 4);
  float*    rmaxv  = (float*)carve((size_t)kB * kN * 4);
  float*    rinvv  = (float*)carve((size_t)kB * kN * 4);
  _Float16* t1h    = (_Float16*)carve((size_t)kB * kCI * kN * 2);
  _Float16* t2h    = (_Float16*)carve((size_t)kB * kCI * kN * 2);
  _Float16* y1h    = (_Float16*)carve((size_t)kB * kCI * kN * 2);
  _Float16* y2h    = (_Float16*)carve((size_t)kB * kCI * kN * 2);

  fold_pre_kernel<<<dim3(2), dim3(256), 0, stream>>>(bn_pre, w_pre, b_pre, WpreH, bpreF);
  fold_post_kernel<<<dim3(2), dim3(256), 0, stream>>>(bn_post, w_post, b_post, WpostH, bpostF);

  const int nel = kB * kC * kN;
  f2h_kernel<<<dim3((nel + 255) / 256), dim3(256), 0, stream>>>(x1, xh, nel);
  f2h_kernel<<<dim3((nel + 255) / 256), dim3(256), 0, stream>>>(x2, xh + (long)nel, nel);

  // Pre GEMMs: Y[i] = Wpre'[i] @ x + b'   (M=128, K=256, N=2304; f16 out)
  const _Float16* xsrc[4] = { xh, xh + (long)nel, xh, xh + (long)nel };  // g1,g2,theta,phi
  for (int i = 0; i < 4; ++i) {
    gemm_wmma_f16<0, 0, 0><<<dim3(kN / 128, 1, kB), 256, 0, stream>>>(
        WpreH + (long)i * kCI * kC, 0, kC,
        xsrc[i], (long)kC * kN, kN,
        nullptr, nullptr, bpreF + i * kCI, nullptr, 0,
        Yh + (long)i * kBS, (long)kCI * kN, kN,
        kN, kC);
  }
  const _Float16* g11 = Yh;
  const _Float16* g21 = Yh + kBS;
  const _Float16* th  = Yh + 2 * kBS;
  const _Float16* ph  = Yh + 3 * kBS;

  // e_time[c][d] = sum_n th[c][n]*ph[d][n]  (B transposed-f16 mode; f32 out)
  gemm_wmma_f16<0, 3, 1><<<dim3(1, 1, kB), 256, 0, stream>>>(
      th, (long)kCI * kN, kN,
      ph, (long)kCI * kN, kN,
      nullptr, nullptr, nullptr, nullptr, 0,
      Et, (long)kCI * kCI, kCI,
      kCI, kN);

  et_softmax_kernel<<<dim3(kCI, kB), 128, 0, stream>>>(Et, EG1h, EG2h);

  // e_space[n][m] = sum_c th[c][n]*ph[c][m]  (A transposed; f32 out)
  gemm_wmma_f16<1, 0, 1><<<dim3(kN / 128, kN / 128, kB), 256, 0, stream>>>(
      th, (long)kCI * kN, kN,
      ph, (long)kCI * kN, kN,
      nullptr, nullptr, nullptr, nullptr, 0,
      Es, (long)kN * kN, kN,
      kN, kCI);

  es_colstats_kernel<<<dim3(kN / 256, 1, kB), 256, 0, stream>>>(Es, cmaxv, cinvv);
  es_rowstats_kernel<<<dim3(kN, 1, kB), 256, 0, stream>>>(Es, rmaxv, rinvv);

  // t1 = et2s @ g11, t2 = et1s @ g21  (M=128, K=128, N=2304; f16 out)
  gemm_wmma_f16<0, 0, 0><<<dim3(kN / 128, 1, kB), 256, 0, stream>>>(
      EG1h, (long)kCI * kCI, kCI, g11, (long)kCI * kN, kN,
      nullptr, nullptr, nullptr, nullptr, 0,
      t1h, (long)kCI * kN, kN, kN, kCI);
  gemm_wmma_f16<0, 0, 0><<<dim3(kN / 128, 1, kB), 256, 0, stream>>>(
      EG2h, (long)kCI * kCI, kCI, g21, (long)kCI * kN, kN,
      nullptr, nullptr, nullptr, nullptr, 0,
      t2h, (long)kCI * kN, kN, kN, kCI);

  // y1 = t1 @ colsoftmax(Es), y2 = t2 @ rowsoftmax(Es)^T (softmax fused in B load)
  gemm_wmma_f16<0, 1, 0><<<dim3(kN / 128, 1, kB), 256, 0, stream>>>(
      t1h, (long)kCI * kN, kN,
      Es, (long)kN * kN, kN,
      cmaxv, cinvv, nullptr, nullptr, 0,
      y1h, (long)kCI * kN, kN, kN, kN);
  gemm_wmma_f16<0, 2, 0><<<dim3(kN / 128, 1, kB), 256, 0, stream>>>(
      t2h, (long)kCI * kN, kN,
      Es, (long)kN * kN, kN,
      rmaxv, rinvv, nullptr, nullptr, 0,
      y2h, (long)kCI * kN, kN, kN, kN);

  // post convs + folded BN + residual -> d_out (out1 then out2)
  gemm_wmma_f16<0, 0, 2><<<dim3(kN / 128, kC / 128, kB), 256, 0, stream>>>(
      WpostH, 0, kCI,
      y1h, (long)kCI * kN, kN,
      nullptr, nullptr, bpostF, x1, (long)kC * kN,
      out, (long)kC * kN, kN, kN, kCI);
  gemm_wmma_f16<0, 0, 2><<<dim3(kN / 128, kC / 128, kB), 256, 0, stream>>>(
      WpostH + (long)kC * kCI, 0, kCI,
      y2h, (long)kCI * kN, kN,
      nullptr, nullptr, bpostF + kC, x2, (long)kC * kN,
      out + (long)kB * kC * kN, (long)kC * kN, kN, kN, kCI);
}
